// ResidualAttentionBlock_91285234909528
// MI455X (gfx1250) — matrix operand order repaired and checked by reading, not compiled, not run on hardware
//
#include <hip/hip_runtime.h>
#include <hip/hip_bf16.h>
#include <math.h>

// ---------------------------------------------------------------------------
// ResidualAttentionBlock for MI455X (gfx1250, wave32, WMMA).
// All GEMMs run through v_wmma_f32_16x16x32_bf16 (bf16 in, f32 accumulate).
// Weights are pre-transposed to [N,K] so every WMMA B-fragment is two
// contiguous 16B loads. GEMM tiles are staged into LDS with
// GLOBAL_LOAD_ASYNC_TO_LDS_B128 (ASYNCcnt) instead of load+ds_store.
// Attention is flash-style (online softmax); V is produced transposed so
// P*V fragments are contiguous global b128 loads.
// ---------------------------------------------------------------------------

typedef __bf16 bf16;
typedef __attribute__((ext_vector_type(16))) __bf16 v16bf;
typedef __attribute__((ext_vector_type(8)))  float  v8f;

#define B_   4
#define S_   1024
#define T_   1500
#define D_   1024
#define H_   16
#define DH_  64
#define DFF_ 4096
#define TKP_ 1504   // T padded to 16 for transposed-V row stride

__device__ __forceinline__ v8f bf16_wmma(v16bf a, v16bf b, v8f c) {
    return __builtin_amdgcn_wmma_f32_16x16x32_bf16(
        false, a, false, b, (short)0, c, false, false);
}

// Async 16B global -> LDS copy (per lane). LDS byte address = low 32 bits of
// the generic pointer to the __shared__ destination (AMDGCN addrspace cast).
__device__ __forceinline__ void async_copy16(const bf16* gsrc, bf16* ldst) {
    unsigned lds = (unsigned)(unsigned long long)ldst;
    asm volatile("global_load_async_to_lds_b128 %0, %1, off"
                 :: "v"(lds), "v"(gsrc) : "memory");
}
__device__ __forceinline__ void async_wait0() {
    asm volatile("s_wait_asynccnt 0x0" ::: "memory");
}

// Load a 16-element bf16 fragment as two 16B vectors from one base (B layout:
// element i <-> K = base + i, contiguous).
__device__ __forceinline__ v16bf load16(const bf16* p) {
    union { uint4 u[2]; v16bf v; } f;
    f.u[0] = *(const uint4*)(p);
    f.u[1] = *(const uint4*)(p + 8);
    return f.v;
}
// A layout: element i <-> K = koff + 16*half + i (contiguous, matches B).
//
__device__ __forceinline__ v16bf load16_a(const bf16* row, int half, int koff) {
    union { uint4 u[2]; v16bf v; } f;
    f.u[0] = *(const uint4*)(row + koff + 16 * half);
    f.u[1] = *(const uint4*)(row + koff + 16 * half + 8);
    return f.v;
}
__device__ __forceinline__ v16bf zero16() {
    union { uint4 u[2]; v16bf v; } f;
    f.u[0] = make_uint4(0u, 0u, 0u, 0u);
    f.u[1] = make_uint4(0u, 0u, 0u, 0u);
    return f.v;
}

// ---------------------------------------------------------------------------
// f32 -> bf16 conversion (no transpose)
// ---------------------------------------------------------------------------
__global__ void cvt_f32_bf16(const float* __restrict__ in, bf16* __restrict__ out,
                             size_t n) {
    size_t i = (size_t)blockIdx.x * blockDim.x + threadIdx.x;
    if (i < n) out[i] = (bf16)in[i];
}

// ---------------------------------------------------------------------------
// f32 [K,N] -> bf16 [N,K] tiled transpose-convert (32x32 tiles via LDS).
// K, N multiples of 32.
// ---------------------------------------------------------------------------
__global__ __launch_bounds__(256)
void cvt_transpose_bf16(const float* __restrict__ in, bf16* __restrict__ out,
                        int K, int N) {
    __shared__ float tile[32][33];
    const int n0 = blockIdx.x * 32;
    const int k0 = blockIdx.y * 32;
    const int tx = threadIdx.x & 31;
    const int ty = threadIdx.x >> 5;   // 8 rows per pass
    for (int i = ty; i < 32; i += 8)
        tile[i][tx] = in[(size_t)(k0 + i) * N + n0 + tx];
    __syncthreads();
    for (int i = ty; i < 32; i += 8)
        out[(size_t)(n0 + i) * K + k0 + tx] = (bf16)tile[tx][i];
}

// ---------------------------------------------------------------------------
// LayerNorm (f32 in) -> bf16 out. One block per row of length D.
// ---------------------------------------------------------------------------
__global__ __launch_bounds__(256)
void layernorm_bf16(const float* __restrict__ x, const float* __restrict__ g,
                    const float* __restrict__ b, bf16* __restrict__ y, int Dd) {
    __shared__ float red1[256];
    __shared__ float red2[256];
    const int row = blockIdx.x;
    const float* xr = x + (size_t)row * Dd;
    float s = 0.f, sq = 0.f;
    for (int i = threadIdx.x; i < Dd; i += 256) {
        float v = xr[i]; s += v; sq += v * v;
    }
    red1[threadIdx.x] = s; red2[threadIdx.x] = sq;
    __syncthreads();
    for (int off = 128; off > 0; off >>= 1) {
        if (threadIdx.x < off) {
            red1[threadIdx.x] += red1[threadIdx.x + off];
            red2[threadIdx.x] += red2[threadIdx.x + off];
        }
        __syncthreads();
    }
    float mean = red1[0] / (float)Dd;
    float var  = red2[0] / (float)Dd - mean * mean;
    float rstd = rsqrtf(var + 1e-5f);
    bf16* yr = y + (size_t)row * Dd;
    for (int i = threadIdx.x; i < Dd; i += 256)
        yr[i] = (bf16)((xr[i] - mean) * rstd * g[i] + b[i]);
}

// ---------------------------------------------------------------------------
// Tiled WMMA GEMM: C[M,N] = A[M,K](bf16) x Bt[N,K](bf16, pre-transposed)
// Block: 128 threads (4 waves). Block tile 64x64, wave tile 32x32 (2x2 WMMA).
// Tiles staged via async global->LDS b128 (ASYNCcnt); fragments are
// ds_load_b128 pairs (padded strides keep 16B alignment).
// EPI: 0=none  1=+bias  2=+bias,gelu(exact)  3=+bias,+resid
// OUT_TRANS: store bf16 transposed per batch: out[(b*N+n)*t_skp + s],
//            where gm = b*t_sk + s  (used for the V projection).
// ---------------------------------------------------------------------------
template <int EPI, bool OUT_BF16, bool OUT_TRANS>
__global__ __launch_bounds__(128)
void gemm_bf16(const bf16* __restrict__ A, const bf16* __restrict__ Bt,
               const float* __restrict__ bias, const float* __restrict__ resid,
               void* __restrict__ Cout, int M, int N, int K,
               int t_sk, int t_skp) {
    constexpr int LDT = 32 + 8;   // 80B row stride (16B multiple)
    __shared__ bf16 As[64 * LDT];
    __shared__ bf16 Bs[64 * LDT];

    const int tid  = threadIdx.x;
    const int lane = tid & 31;
    const int wave = tid >> 5;
    const int m0 = blockIdx.y * 64;
    const int n0 = blockIdx.x * 64;
    const int wm = (wave >> 1) * 32;
    const int wn = (wave & 1) * 32;
    const int l15  = lane & 15;
    const int half = lane >> 4;

    v8f acc[2][2];
    #pragma unroll
    for (int i = 0; i < 2; ++i)
        #pragma unroll
        for (int j = 0; j < 2; ++j)
            #pragma unroll
            for (int r = 0; r < 8; ++r) acc[i][j][r] = 0.f;

    // Per-thread staging coordinates (64 rows x 32 cols, 8 elems per slot)
    const int srow = tid >> 2;
    const int scol = (tid & 3) * 8;

    for (int k0 = 0; k0 < K; k0 += 32) {
        __syncthreads();   // previous iteration's fragment reads done (WAR)
        // Stage A tile 64x32 async (row-guarded for M tails; unloaded rows
        // only feed accumulator rows that are store-guarded in the epilogue)
        #pragma unroll
        for (int it = 0; it < 2; ++it) {
            int row = srow + it * 32;
            int gm = m0 + row;
            if (gm < M) {
                const bf16* src = A + (size_t)gm * K + k0 + scol;
                async_copy16(src, &As[row * LDT + scol]);
                __builtin_prefetch(src + 32, 0, 1);   // next K tile
            }
        }
        // Stage Bt tile 64(n) x 32(k) async
        #pragma unroll
        for (int it = 0; it < 2; ++it) {
            int row = srow + it * 32;
            const bf16* src = Bt + (size_t)(n0 + row) * K + k0 + scol;
            async_copy16(src, &Bs[row * LDT + scol]);
            __builtin_prefetch(src + 32, 0, 1);       // next K tile
        }
        async_wait0();     // own wave's async copies landed in LDS
        __syncthreads();   // all waves' copies visible

        v16bf afrag[2], bfrag[2];
        #pragma unroll
        for (int tm = 0; tm < 2; ++tm)
            afrag[tm] = load16_a(&As[(wm + tm * 16 + l15) * LDT], half, 0);
        #pragma unroll
        for (int tn = 0; tn < 2; ++tn)
            bfrag[tn] = load16(&Bs[(wn + tn * 16 + l15) * LDT + 16 * half]);
        #pragma unroll
        for (int tm = 0; tm < 2; ++tm)
            #pragma unroll
            for (int tn = 0; tn < 2; ++tn)
                acc[tm][tn] = bf16_wmma(afrag[tm], bfrag[tn], acc[tm][tn]);
    }

    // Epilogue
    #pragma unroll
    for (int tm = 0; tm < 2; ++tm) {
        #pragma unroll
        for (int tn = 0; tn < 2; ++tn) {
            int gn = n0 + wn + tn * 16 + l15;
            float bv = (EPI != 0) ? bias[gn] : 0.0f;
            #pragma unroll
            for (int r = 0; r < 8; ++r) {
                int gm = m0 + wm + tm * 16 + r + 8 * half;
                if (gm >= M) continue;
                float val = acc[tm][tn][r] + bv;
                if (EPI == 2) {
                    val = 0.5f * val * (1.0f + erff(val * 0.70710678118654752f));
                } else if (EPI == 3) {
                    val += resid[(size_t)gm * N + gn];
                }
                if (OUT_TRANS) {
                    int bidx = gm / t_sk;
                    int sidx = gm - bidx * t_sk;
                    ((bf16*)Cout)[((size_t)bidx * N + gn) * t_skp + sidx] = (bf16)val;
                } else if (OUT_BF16) {
                    ((bf16*)Cout)[(size_t)gm * N + gn] = (bf16)val;
                } else {
                    ((float*)Cout)[(size_t)gm * N + gn] = val;
                }
            }
        }
    }
}

// ---------------------------------------------------------------------------
// Flash attention: one wave handles 16 queries x DH=64 for one (b,h).
// 32 keys per step: QK^T (4 WMMA), online softmax (shfl_xor row reductions),
// probs C-layout -> A-layout via LDS, P*V (4 WMMA, V pre-transposed).
// Q,K: bf16 [rows, D] (head offset h*DH).  Vt: bf16 [B][D][Skp].
// Tail keys carry prob == 0 (masked before exp), so V padding is harmless.
// ---------------------------------------------------------------------------
template <bool CAUSAL>
__global__ __launch_bounds__(32)
void flash_attn(const bf16* __restrict__ Q, const bf16* __restrict__ K,
                const bf16* __restrict__ Vt, bf16* __restrict__ O,
                int Sq, int Sk, int Skp, float scale) {
    __shared__ bf16 pLds[16 * 40];
    const int lane = threadIdx.x;
    const int l15  = lane & 15;
    const int half = lane >> 4;
    const int qm = blockIdx.x * 16;
    const int bh = blockIdx.y;
    const int b = bh / H_;
    const int h = bh % H_;
    const bf16* Qb = Q + ((size_t)b * Sq) * D_ + h * DH_;
    const bf16* Kb = K + ((size_t)b * Sk) * D_ + h * DH_;
    bf16*       Ob = O + ((size_t)b * Sq) * D_ + h * DH_;

    // Q fragments: d-steps 0..31 and 32..63 (A layout, contiguous halves)
    const bf16* qrow = Qb + (size_t)(qm + l15) * D_;
    v16bf aq[2];
    #pragma unroll
    for (int d0 = 0; d0 < 2; ++d0)
        aq[d0] = load16_a(qrow, half, d0 * 32);

    // Per-dt transposed-V row base: lane column = output d index
    const bf16* vrow[4];
    #pragma unroll
    for (int dt = 0; dt < 4; ++dt)
        vrow[dt] = Vt + ((size_t)b * D_ + h * DH_ + dt * 16 + l15) * Skp + 16 * half;

    v8f acc[4];
    #pragma unroll
    for (int d = 0; d < 4; ++d)
        #pragma unroll
        for (int r = 0; r < 8; ++r) acc[d][r] = 0.f;
    float mrow[8], lrow[8];
    #pragma unroll
    for (int r = 0; r < 8; ++r) { mrow[r] = -1e30f; lrow[r] = 0.f; }

    const int kend = CAUSAL ? (qm + 16) : Sk;
    for (int kt = 0; kt < kend; kt += 32) {
        // --- scores: 16 x 32 as two 16x16 WMMA tiles ---
        v8f sc[2];
        #pragma unroll
        for (int nt = 0; nt < 2; ++nt) {
            int kk = kt + nt * 16 + l15;
            v16bf bk0, bk1;
            if (kk < Sk) {
                const bf16* krow = Kb + (size_t)kk * D_ + 16 * half;
                bk0 = load16(krow);
                bk1 = load16(krow + 32);
            } else {
                bk0 = zero16();
                bk1 = zero16();
            }
            v8f c;
            #pragma unroll
            for (int r = 0; r < 8; ++r) c[r] = 0.f;
            c = bf16_wmma(aq[0], bk0, c);
            c = bf16_wmma(aq[1], bk1, c);
            sc[nt] = c;
        }
        // --- online softmax update ---
        float pnew[2][8];
        #pragma unroll
        for (int r = 0; r < 8; ++r) {
            int qg  = qm + r + 8 * half;
            int k0g = kt + l15;
            int k1g = kt + 16 + l15;
            float s0 = sc[0][r] * scale;
            float s1 = sc[1][r] * scale;
            if ((CAUSAL && k0g > qg) || k0g >= Sk) s0 = -1e30f;
            if ((CAUSAL && k1g > qg) || k1g >= Sk) s1 = -1e30f;
            float mx = fmaxf(s0, s1);
            #pragma unroll
            for (int off = 1; off < 16; off <<= 1)
                mx = fmaxf(mx, __shfl_xor(mx, off, 32));
            float mn = fmaxf(mrow[r], mx);
            float p0 = __expf(s0 - mn);
            float p1 = __expf(s1 - mn);
            float corr = __expf(mrow[r] - mn);
            float ps = p0 + p1;
            #pragma unroll
            for (int off = 1; off < 16; off <<= 1)
                ps += __shfl_xor(ps, off, 32);
            lrow[r] = lrow[r] * corr + ps;
            mrow[r] = mn;
            #pragma unroll
            for (int d = 0; d < 4; ++d) acc[d][r] *= corr;
            pnew[0][r] = p0; pnew[1][r] = p1;
        }
        // --- transpose probs (C layout) -> A layout via LDS ---
        __syncthreads();
        #pragma unroll
        for (int r = 0; r < 8; ++r) {
            int mloc = r + 8 * half;
            pLds[mloc * 40 + l15]      = (bf16)pnew[0][r];
            pLds[mloc * 40 + 16 + l15] = (bf16)pnew[1][r];
        }
        __syncthreads();
        v16bf pf = load16_a(&pLds[l15 * 40], half, 0);
        // --- P x V (transposed V: contiguous K per lane) ---
        #pragma unroll
        for (int dt = 0; dt < 4; ++dt)
            acc[dt] = bf16_wmma(pf, load16(vrow[dt] + kt), acc[dt]);
    }
    // --- finalize ---
    #pragma unroll
    for (int r = 0; r < 8; ++r) {
        float inv = 1.0f / lrow[r];
        int sr = qm + r + 8 * half;
        #pragma unroll
        for (int dt = 0; dt < 4; ++dt)
            Ob[(size_t)sr * D_ + dt * 16 + l15] = (bf16)(acc[dt][r] * inv);
    }
}

// ---------------------------------------------------------------------------
// Orchestration
// ---------------------------------------------------------------------------
extern "C" void kernel_launch(void* const* d_in, const int* in_sizes, int n_in,
                              void* d_out, int out_size, void* d_ws, size_t ws_size,
                              hipStream_t stream) {
    const int MS = B_ * S_;   // 4096 query rows
    const int MT = B_ * T_;   // 6000 memory rows

    const float* x         = (const float*)d_in[0];
    const float* xa        = (const float*)d_in[1];
    // d_in[2] = mask (causal triu -1e9): implemented analytically
    const float* attn_ln_g = (const float*)d_in[3];
    const float* attn_ln_b = (const float*)d_in[4];
    const float* sa_wq = (const float*)d_in[5];
    const float* sa_bq = (const float*)d_in[6];
    const float* sa_wk = (const float*)d_in[7];
    const float* sa_wv = (const float*)d_in[8];
    const float* sa_bv = (const float*)d_in[9];
    const float* sa_wo = (const float*)d_in[10];
    const float* sa_bo = (const float*)d_in[11];
    const float* cross_ln_g = (const float*)d_in[12];
    const float* cross_ln_b = (const float*)d_in[13];
    const float* ca_wq = (const float*)d_in[14];
    const float* ca_bq = (const float*)d_in[15];
    const float* ca_wk = (const float*)d_in[16];
    const float* ca_wv = (const float*)d_in[17];
    const float* ca_bv = (const float*)d_in[18];
    const float* ca_wo = (const float*)d_in[19];
    const float* ca_bo = (const float*)d_in[20];
    const float* mlp_ln_g = (const float*)d_in[21];
    const float* mlp_ln_b = (const float*)d_in[22];
    const float* mlp_w1 = (const float*)d_in[23];
    const float* mlp_b1 = (const float*)d_in[24];
    const float* mlp_w2 = (const float*)d_in[25];
    const float* mlp_b2 = (const float*)d_in[26];

    // Workspace carve (256B aligned)
    char* wsp = (char*)d_ws;
    auto carve = [&](size_t bytes) -> void* {
        void* p = (void*)wsp;
        wsp += (bytes + 255) & ~(size_t)255;
        return p;
    };
    const size_t DD = (size_t)D_ * D_;
    bf16* w_sq = (bf16*)carve(DD * 2);
    bf16* w_sk = (bf16*)carve(DD * 2);
    bf16* w_sv = (bf16*)carve(DD * 2);
    bf16* w_so = (bf16*)carve(DD * 2);
    bf16* w_cq = (bf16*)carve(DD * 2);
    bf16* w_ck = (bf16*)carve(DD * 2);
    bf16* w_cv = (bf16*)carve(DD * 2);
    bf16* w_co = (bf16*)carve(DD * 2);
    bf16* w_m1 = (bf16*)carve((size_t)D_ * DFF_ * 2);
    bf16* w_m2 = (bf16*)carve((size_t)DFF_ * D_ * 2);
    bf16* xln  = (bf16*)carve((size_t)MS * D_ * 2);
    bf16* xab  = (bf16*)carve((size_t)MT * D_ * 2);
    bf16* qb   = (bf16*)carve((size_t)MS * D_ * 2);
    bf16* kb   = (bf16*)carve((size_t)MT * D_ * 2);
    bf16* vtb  = (bf16*)carve((size_t)B_ * D_ * TKP_ * 2);   // transposed V
    bf16* ab   = (bf16*)carve((size_t)MS * D_ * 2);
    float* res = (float*)carve((size_t)MS * D_ * 4);
    bf16* hb   = (bf16*)carve((size_t)MS * DFF_ * 2);

    auto cvt = [&](const float* src, bf16* dst, size_t n) {
        cvt_f32_bf16<<<dim3((unsigned)((n + 255) / 256)), dim3(256), 0, stream>>>(src, dst, n);
    };
    // Weights -> bf16, transposed to [N,K]
    auto cvt_t = [&](const float* src, bf16* dst, int K, int N) {
        cvt_transpose_bf16<<<dim3(N / 32, K / 32), dim3(256), 0, stream>>>(src, dst, K, N);
    };
    cvt_t(sa_wq, w_sq, D_, D_); cvt_t(sa_wk, w_sk, D_, D_);
    cvt_t(sa_wv, w_sv, D_, D_); cvt_t(sa_wo, w_so, D_, D_);
    cvt_t(ca_wq, w_cq, D_, D_); cvt_t(ca_wk, w_ck, D_, D_);
    cvt_t(ca_wv, w_cv, D_, D_); cvt_t(ca_wo, w_co, D_, D_);
    cvt_t(mlp_w1, w_m1, D_, DFF_);
    cvt_t(mlp_w2, w_m2, DFF_, D_);
    cvt(xa, xab, (size_t)MT * D_);

    auto ggrid = [](int M, int N) { return dim3((unsigned)(N / 64), (unsigned)((M + 63) / 64)); };
    const dim3 gblk(128);
    const float scale = 0.125f;   // DH^-0.5

    // ---- self-attention ----
    layernorm_bf16<<<dim3(MS), dim3(256), 0, stream>>>(x, attn_ln_g, attn_ln_b, xln, D_);
    gemm_bf16<1, true, false><<<ggrid(MS, D_), gblk, 0, stream>>>(xln, w_sq, sa_bq, nullptr, qb, MS, D_, D_, 0, 0);
    gemm_bf16<0, true, false><<<ggrid(MS, D_), gblk, 0, stream>>>(xln, w_sk, nullptr, nullptr, kb, MS, D_, D_, 0, 0);
    gemm_bf16<1, true, true ><<<ggrid(MS, D_), gblk, 0, stream>>>(xln, w_sv, sa_bv, nullptr, vtb, MS, D_, D_, S_, S_);
    flash_attn<true><<<dim3(S_ / 16, B_ * H_), dim3(32), 0, stream>>>(qb, kb, vtb, ab, S_, S_, S_, scale);
    gemm_bf16<3, false, false><<<ggrid(MS, D_), gblk, 0, stream>>>(ab, w_so, sa_bo, x, res, MS, D_, D_, 0, 0);

    // ---- cross-attention ----
    layernorm_bf16<<<dim3(MS), dim3(256), 0, stream>>>(res, cross_ln_g, cross_ln_b, xln, D_);
    gemm_bf16<1, true, false><<<ggrid(MS, D_), gblk, 0, stream>>>(xln, w_cq, ca_bq, nullptr, qb, MS, D_, D_, 0, 0);
    gemm_bf16<0, true, false><<<ggrid(MT, D_), gblk, 0, stream>>>(xab, w_ck, nullptr, nullptr, kb, MT, D_, D_, 0, 0);
    gemm_bf16<1, true, true ><<<ggrid(MT, D_), gblk, 0, stream>>>(xab, w_cv, ca_bv, nullptr, vtb, MT, D_, D_, T_, TKP_);
    flash_attn<false><<<dim3(S_ / 16, B_ * H_), dim3(32), 0, stream>>>(qb, kb, vtb, ab, S_, T_, TKP_, scale);
    gemm_bf16<3, false, false><<<ggrid(MS, D_), gblk, 0, stream>>>(ab, w_co, ca_bo, res, res, MS, D_, D_, 0, 0);

    // ---- MLP ----
    layernorm_bf16<<<dim3(MS), dim3(256), 0, stream>>>(res, mlp_ln_g, mlp_ln_b, xln, D_);
    gemm_bf16<2, true, false><<<ggrid(MS, DFF_), gblk, 0, stream>>>(xln, w_m1, mlp_b1, nullptr, hb, MS, DFF_, D_, 0, 0);
    gemm_bf16<3, false, false><<<ggrid(MS, D_), gblk, 0, stream>>>(hb, w_m2, mlp_b2, res, (float*)d_out, MS, D_, DFF_, 0, 0);

    (void)in_sizes; (void)n_in; (void)out_size; (void)ws_size;
}
